// VectorQuantizer_30339648979547
// MI455X (gfx1250) — matrix operand order repaired and checked by reading, not compiled, not run on hardware
//
#include <hip/hip_runtime.h>
#include <hip/hip_bf16.h>

// ---------------------------------------------------------------------------
// VQ-VAE vector quantizer for MI455X (gfx1250, wave32, WMMA + async-to-LDS).
//
// argmin_k dist[n,k] only needs s = ||e_k||^2 - 2 x_n.e_k; the hot loop is the
// [131072,256]x[256,1024] GEMM via v_wmma_f32_16x16x32_f16 (f16 in, f32 acc).
//
// Roofline: 68.7 GFLOP of WMMA vs 256MB HBM (11us @ 23.3TB/s). The codebook
// (512KB f16) is staged once per 128-row workgroup into LDS by the async DMA
// engine (triple-buffered ring, prefetch depth 2, ASYNCcnt) -> 512MB L2 total,
// balanced with the WMMA time. A fragments live in VGPRs for the whole kernel
// (each fragment feeds 64 WMMAs).
// ---------------------------------------------------------------------------

typedef __attribute__((ext_vector_type(16))) _Float16 v16h;
typedef __attribute__((ext_vector_type(8)))  _Float16 v8h;
typedef __attribute__((ext_vector_type(8)))  float    v8f;

#define K_CODES 1024
#define DIM     256
#define KT      (K_CODES / 16)    // 64 code tiles
#define DT      (DIM / 32)        // 8 K-steps of 32
#define M_TILE  128               // rows per workgroup
#define BLOCK   256               // 8 waves x 16 rows
#define ROWS_PASS 64              // x-tile staged in two passes of 64 rows
#define LDS_ROW (DIM + 8)         // halves; pad -> 528B row stride, bank-spread
#define HW      4096              // 64*64
#define N_ROWS  (32 * HW)         // 131072

#define BUF_HALVES (DT * 32 * 16) // 4096 halves = 8KB per code-tile of B
#define BUF_BYTES  (BUF_HALVES * 2)

// ws layout: [0, 512KB) packed f16 B-fragments; then float enorm2[1024]
#define BFRAG_HALVES ((size_t)KT * BUF_HALVES)

// ---------------------------------------------------------------------------
// Async DMA helpers (GVS: global = SADDR + VADDR32 + offset; LDS = VDST32+off)
// ---------------------------------------------------------------------------
__device__ __forceinline__ void asyncCopy32B(unsigned ldsByte, unsigned voff,
                                             const void* sbase) {
    asm volatile(
        "global_load_async_to_lds_b128 %0, %1, %2 offset:0\n\t"
        "global_load_async_to_lds_b128 %0, %1, %2 offset:16"
        :: "v"(ldsByte), "v"(voff), "s"(sbase)
        : "memory");
}
__device__ __forceinline__ void asyncCopy16B(unsigned ldsByte, unsigned voff,
                                             const void* sbase) {
    asm volatile(
        "global_load_async_to_lds_b128 %0, %1, %2 offset:0"
        :: "v"(ldsByte), "v"(voff), "s"(sbase)
        : "memory");
}
__device__ __forceinline__ void waitAsync2() {
    asm volatile("s_wait_asynccnt 0x2" ::: "memory");
}

// argmin step over XOR partner within 32 lanes (ds_swizzle, immediate mode)
template <int IMM>
__device__ __forceinline__ void minStep(float& s, int& id) {
    float so = __int_as_float(__builtin_amdgcn_ds_swizzle(__float_as_int(s), IMM));
    int   io = __builtin_amdgcn_ds_swizzle(id, IMM);
    if (so < s || (so == s && io < id)) { s = so; id = io; }
}
#define SWZ_XOR(m) (((m) << 10) | 0x1F)

// ---------------------------------------------------------------------------
// Kernel 1: pack codebook into WMMA B-fragment order (f16).
// B 32x16 layout used here (contiguous k, same order as the A frags):
//   lane&15 = column (code); lanes 0-15 hold K=0..15, lanes 16-31 K=16..31.
// ---------------------------------------------------------------------------
__global__ void vq_pack_b(const float* __restrict__ emb, _Float16* __restrict__ bfrag) {
    const int frag = blockIdx.x;          // kt*8 + dt, 0..511
    const int lane = threadIdx.x;         // 0..31
    const int kt = frag >> 3;
    const int dt = frag & 7;
    const int col = kt * 16 + (lane & 15);
    const int kbase = dt * 32 + ((lane & 16) ? 16 : 0);
    _Float16* dst = bfrag + ((size_t)frag * 32 + lane) * 16;
    #pragma unroll
    for (int h = 0; h < 16; ++h)
        dst[h] = (_Float16)emb[(size_t)col * DIM + kbase + h];
}

// ---------------------------------------------------------------------------
// Kernel 2: ||e_k||^2 per code, and zero the loss accumulator.
// ---------------------------------------------------------------------------
__global__ void vq_enorm(const float* __restrict__ emb, float* __restrict__ enorm2,
                         float* __restrict__ d_loss) {
    const int k = blockIdx.x * blockDim.x + threadIdx.x;
    if (k == 0) *d_loss = 0.0f;
    if (k < K_CODES) {
        float s = 0.f;
        #pragma unroll 4
        for (int d = 0; d < DIM; ++d) {
            float v = emb[(size_t)k * DIM + d];
            s += v * v;
        }
        enorm2[k] = s;
    }
}

// ---------------------------------------------------------------------------
// Kernel 3: main. 128 rows/WG, 8 waves x 16 rows; B streamed to LDS by DMA.
// ---------------------------------------------------------------------------
__global__ __launch_bounds__(BLOCK)
void vq_main(const float* __restrict__ in, const float* __restrict__ emb,
             const _Float16* __restrict__ bfrag, const float* __restrict__ enorm2,
             const float* __restrict__ beta_p, float* __restrict__ d_loss,
             float* __restrict__ outq) {
    __shared__ alignas(16) _Float16 xs[ROWS_PASS * LDS_ROW];   // ~33KB, reused 2x
    __shared__ alignas(32) _Float16 bufB[3 * BUF_HALVES];      // 24KB ring
    __shared__ alignas(16) float enormS[K_CODES];              // 4KB
    __shared__ float xnorm2[M_TILE];
    __shared__ float rowScore[M_TILE];
    __shared__ int   rowIdx[M_TILE];

    const int tid  = threadIdx.x;
    const int lane = tid & 31;
    const int wave = tid >> 5;                       // 0..7
    const int colLane = lane & 15;
    const int hiHalf  = (lane >> 4) & 1;
    const long n0  = (long)blockIdx.x * M_TILE;
    const int  b   = (int)(n0 >> 12);
    const int  hw0 = (int)(n0 & (HW - 1));
    const float* inb = in + (size_t)b * DIM * HW;    // in[b, d, hw]

    // ---- kick off async DMA (enorm table + B tiles 0,1) while we stage x ----
    const unsigned bufBase = (unsigned)(uintptr_t)&bufB[0];
    const unsigned enBase  = (unsigned)(uintptr_t)&enormS[0];
    const unsigned myOff   = (unsigned)tid * 32;     // 256 thr * 32B = 8KB tile
    asyncCopy16B(enBase + tid * 16, (unsigned)tid * 16, enorm2);   // 4KB table
    asyncCopy32B(bufBase + 0 * BUF_BYTES + myOff, myOff, bfrag + 0 * BUF_HALVES);
    asyncCopy32B(bufBase + 1 * BUF_BYTES + myOff, myOff, bfrag + 1 * BUF_HALVES);

    if (tid < M_TILE) xnorm2[tid] = 0.f;

    // ---- stage x tile through LDS in two 64-row passes; build A frags ----
    // A layout (contiguous k, same order as packed B): lane&15 = M row;
    //   lanes 0-15: K = 0..15 ; lanes 16-31: K = 16..31.
    v16h a[DT];
    for (int half = 0; half < 2; ++half) {
        __syncthreads();                             // xs free; xnorm2 zeroed
        const int r0 = half * ROWS_PASS;
        const int r  = tid & (ROWS_PASS - 1);        // each thread: ONE row
        float part = 0.f;
        for (int i = tid; i < ROWS_PASS * DIM; i += BLOCK) {
            int d = i >> 6;
            float v = inb[(size_t)d * HW + hw0 + r0 + r];
            part += v * v;
            xs[r * LDS_ROW + d] = (_Float16)v;
        }
        atomicAdd(&xnorm2[r0 + r], part);            // 4 partials per row
        __syncthreads();
        if ((wave >> 2) == half) {                   // waves 0-3: rows 0-63; 4-7: 64-127
            const _Float16* arow = xs + ((wave & 3) * 16 + colLane) * LDS_ROW;
            #pragma unroll
            for (int dt = 0; dt < DT; ++dt) {
                int d0 = dt * 32 + (hiHalf ? 16 : 0);
                v8h lo = *(const v8h*)(arow + d0);        // K 0..7   (or 16..23)
                v8h hi = *(const v8h*)(arow + d0 + 8);    // K 8..15  (or 24..31)
                v16h av;
                #pragma unroll
                for (int h = 0; h < 8; ++h) { av[h] = lo[h]; av[h + 8] = hi[h]; }
                a[dt] = av;
            }
        }
    }
    waitAsync2();                                    // enorm + B tile 0 landed
    __syncthreads();

    // ---- hot loop: 64 code tiles; ds_load_b128 -> wmma; DMA prefetch kt+2 ----
    float best[8]; int bidx[8];
    #pragma unroll
    for (int r = 0; r < 8; ++r) { best[r] = 3.4e38f; bidx[r] = 0; }

    unsigned rdOff = 0;                              // rotating ring offsets
    unsigned wrOff = 2 * BUF_BYTES;
    for (int kt = 0; kt < KT; ++kt) {
        int pf = kt + 2; if (pf > KT - 1) pf = KT - 1;   // clamped: dead-slot copy
        asyncCopy32B(bufBase + wrOff + myOff, myOff,
                     bfrag + (size_t)pf * BUF_HALVES);

        const _Float16* bl = (const _Float16*)((const char*)bufB + rdOff) + lane * 16;
        v8f c = {};
        v16h bcur = *(const v16h*)bl;                // fragment dt=0
        #pragma unroll
        for (int dt = 0; dt < DT; ++dt) {
            v16h bnext = bcur;
            if (dt + 1 < DT) bnext = *(const v16h*)(bl + (dt + 1) * 512);
            c = __builtin_amdgcn_wmma_f32_16x16x32_f16(
                    false, a[dt], false, bcur, (short)0, c, false, false);
            bcur = bnext;
        }
        const int k = kt * 16 + colLane;             // lane's column = code
        const float en = enormS[k];
        #pragma unroll
        for (int r = 0; r < 8; ++r) {
            float s = en - 2.0f * c[r];
            if (s < best[r]) { best[r] = s; bidx[r] = k; }
        }
        waitAsync2();                                // tile kt+1 complete
        __syncthreads();                             // all waves left buf rdOff
        rdOff += BUF_BYTES; if (rdOff == 3 * BUF_BYTES) rdOff = 0;
        wrOff += BUF_BYTES; if (wrOff == 3 * BUF_BYTES) wrOff = 0;
    }

    // ---- argmin across the 16 lanes of each half (ds_swizzle butterfly) ----
    #pragma unroll
    for (int r = 0; r < 8; ++r) {
        float s = best[r]; int id = bidx[r];
        minStep<SWZ_XOR(1)>(s, id);
        minStep<SWZ_XOR(2)>(s, id);
        minStep<SWZ_XOR(4)>(s, id);
        minStep<SWZ_XOR(8)>(s, id);
        if (colLane == 0) {
            int row = wave * 16 + r + (hiHalf ? 8 : 0);
            rowScore[row] = s;                       // ||e*||^2 - 2 x.e*
            rowIdx[row]   = id;
        }
    }
    __syncthreads();

    // ---- loss: one atomic per workgroup ----
    if (tid == 0) {
        float acc = 0.f;
        for (int r = 0; r < M_TILE; ++r) acc += rowScore[r] + xnorm2[r];
        const float beta = *beta_p;
        const float scale = (1.0f + beta) / (float)((long)N_ROWS * DIM);
        atomicAdd(d_loss, acc * scale);
    }

    // ---- quantized output: out[b, d, hw0+r] = emb[idx[r], d] (coalesced) ----
    float* outb = outq + (size_t)b * DIM * HW + hw0;
    for (int i = tid; i < M_TILE * DIM; i += BLOCK) {
        int r = i & (M_TILE - 1);
        int d = i >> 7;
        outb[(size_t)d * HW + r] = emb[(size_t)rowIdx[r] * DIM + d];
    }
}

// ---------------------------------------------------------------------------
extern "C" void kernel_launch(void* const* d_in, const int* in_sizes, int n_in,
                              void* d_out, int out_size, void* d_ws, size_t ws_size,
                              hipStream_t stream) {
    const float* in   = (const float*)d_in[0];   // [32,256,64,64]
    const float* emb  = (const float*)d_in[1];   // [1024,256]
    const float* beta = (const float*)d_in[2];   // scalar

    float* loss = (float*)d_out;                 // out[0] = loss
    float* outq = (float*)d_out + 1;             // then [32,256,64,64]

    _Float16* bfrag = (_Float16*)d_ws;
    float* enorm2   = (float*)((char*)d_ws + BFRAG_HALVES * sizeof(_Float16));

    vq_pack_b<<<KT * DT, 32, 0, stream>>>(emb, bfrag);
    vq_enorm<<<(K_CODES + 255) / 256, 256, 0, stream>>>(emb, enorm2, loss);
    vq_main<<<N_ROWS / M_TILE, BLOCK, 0, stream>>>(in, emb, bfrag, enorm2,
                                                   beta, loss, outq);
}
